// SelfAttention_38800734552185
// MI455X (gfx1250) — matrix operand order repaired and checked by reading, not compiled, not run on hardware
//
#include <hip/hip_runtime.h>

// ---------------------------------------------------------------------------
// DiT self-attention block for MI455X (gfx1250, wave32, WMMA 16x16x32 bf16).
// Register-blocked WMMA GEMMs + flash attention with TDM-staged KV tiles.
// ---------------------------------------------------------------------------

#define DIM      192
#define HEADS    16
#define DHEAD    64
#define INNER    1024    // HEADS*DHEAD
#define BATCH    8
#define NTOK     1024
#define ROWS     (BATCH * NTOK)   // 8192

typedef __attribute__((ext_vector_type(16))) __bf16 v16bf;
typedef __attribute__((ext_vector_type(8)))  float  v8f;

union FragB16 { v16bf v; unsigned int u[8]; };

__device__ __forceinline__ unsigned short f2bf(float f) {
  union { float f; unsigned u; } x; x.f = f;
  unsigned r = x.u + 0x7FFFu + ((x.u >> 16) & 1u);   // round-to-nearest-even
  return (unsigned short)(r >> 16);
}

// K-pair layout for 16-bit A fragments (contiguous, matches B loads):
// VGPR j: K = 16*hi + 2j
__device__ __forceinline__ int a_kbase(int j, int hi) {
  return 2 * j + 16 * hi;
}

__device__ __forceinline__ float rmax16(float x) {
  x = fmaxf(x, __shfl_xor(x, 1, 16));
  x = fmaxf(x, __shfl_xor(x, 2, 16));
  x = fmaxf(x, __shfl_xor(x, 4, 16));
  x = fmaxf(x, __shfl_xor(x, 8, 16));
  return x;
}
__device__ __forceinline__ float rsum16(float x) {
  x += __shfl_xor(x, 1, 16);
  x += __shfl_xor(x, 2, 16);
  x += __shfl_xor(x, 4, 16);
  x += __shfl_xor(x, 8, 16);
  return x;
}

// ---------------------------------------------------------------------------
// Tensor Data Mover: 2-D tile DMA global -> LDS (D# per ISA 08_async_tensor).
// ---------------------------------------------------------------------------
#if __has_builtin(__builtin_amdgcn_tensor_load_to_lds)
#define USE_TDM 1
typedef __attribute__((ext_vector_type(4))) unsigned int v4u;
typedef __attribute__((ext_vector_type(8))) int v8i_t;
typedef __attribute__((ext_vector_type(4))) int v4i_t;

// Loads a tile of (tile_y rows) x (tile_x elems, contiguous, 2B each) from a
// row-major tensor with row stride `stride0` elems into contiguous LDS.
__device__ __forceinline__ void tdm_load_2d(unsigned lds_byte, unsigned long long ga,
                                            unsigned tile_x, unsigned tile_y,
                                            unsigned tdim0, unsigned tdim1,
                                            unsigned stride0) {
  v4u g0;
  g0[0] = 1u;                                             // count=1 (valid D#)
  g0[1] = lds_byte;                                       // lds_addr
  g0[2] = (unsigned)(ga & 0xFFFFFFFFull);                 // global_addr[31:0]
  g0[3] = ((unsigned)(ga >> 32) & 0x01FFFFFFu) | 0x80000000u; // ga[56:32]|type=2
  v8i_t g1;
  g1[0] = (int)(1u << 16);                                // data_size=1 (2 bytes)
  g1[1] = (int)(tdim0 << 16);                             // tensor_dim0[15:0]@48
  g1[2] = (int)((tdim0 >> 16) | (tdim1 << 16));           // dim0[31:16], dim1[15:0]
  g1[3] = (int)((tdim1 >> 16) | (tile_x << 16));          // dim1[31:16], tile_dim0
  g1[4] = (int)(tile_y & 0xFFFFu);                        // tile_dim1 (tile_dim2=0)
  g1[5] = (int)stride0;                                   // tensor_dim0_stride[31:0]
  g1[6] = 0;
  g1[7] = 0;
  v4i_t z4 = {};
#if defined(__clang_major__) && __clang_major__ >= 23
  v8i_t z8 = {};
  __builtin_amdgcn_tensor_load_to_lds(g0, g1, z4, z4, z8, 0);
#else
  __builtin_amdgcn_tensor_load_to_lds(g0, g1, z4, z4, 0);
#endif
}
#else
#define USE_TDM 0
#endif

// ---------------------------------------------------------------------------
// 1) Weight conversions: transpose to [n][k] bf16 so WMMA B-operand K-pairs
//    are dword-contiguous in memory.
// ---------------------------------------------------------------------------
__global__ void conv_wqkv(const float* __restrict__ w, unsigned short* __restrict__ wT) {
  int i = blockIdx.x * 256 + threadIdx.x;            // over 3*INNER * DIM
  if (i >= 3 * INNER * DIM) return;
  int n = i / DIM, k = i % DIM;
  wT[i] = f2bf(w[k * (3 * INNER) + n]);
}

__global__ void conv_wout(const float* __restrict__ w, unsigned short* __restrict__ wT) {
  int i = blockIdx.x * 256 + threadIdx.x;            // over DIM * INNER
  if (i >= DIM * INNER) return;
  int n = i / INNER, k = i % INNER;
  wT[i] = f2bf(w[k * DIM + n]);
}

// ---------------------------------------------------------------------------
// 2) Fused LayerNorm + adaLN modulation -> bf16 h [ROWS, DIM].
// ---------------------------------------------------------------------------
__global__ __launch_bounds__(256) void ln_mod(
    const float* __restrict__ x, const float* __restrict__ shift,
    const float* __restrict__ scale, const float* __restrict__ nw,
    const float* __restrict__ nb, unsigned short* __restrict__ h) {
  int wave = threadIdx.x >> 5, lane = threadIdx.x & 31;
  int row = blockIdx.x * 8 + wave;                   // 0..ROWS-1
  int bb = row >> 10;
  const float* xr = x + (size_t)row * DIM;
  float v[6];
  float s = 0.f;
#pragma unroll
  for (int i = 0; i < 6; ++i) { v[i] = xr[lane + 32 * i]; s += v[i]; }
#pragma unroll
  for (int m = 1; m < 32; m <<= 1) s += __shfl_xor(s, m, 32);
  float mu = s * (1.f / DIM);
  float vs = 0.f;
#pragma unroll
  for (int i = 0; i < 6; ++i) { float d = v[i] - mu; vs += d * d; }
#pragma unroll
  for (int m = 1; m < 32; m <<= 1) vs += __shfl_xor(vs, m, 32);
  float rsq = rsqrtf(vs * (1.f / DIM) + 1e-5f);
#pragma unroll
  for (int i = 0; i < 6; ++i) {
    int c = lane + 32 * i;
    float y = (v[i] - mu) * rsq * nw[c] + nb[c];
    y = y * (1.f + scale[bb * DIM + c]) + shift[bb * DIM + c];
    h[(size_t)row * DIM + c] = f2bf(y);
  }
}

// ---------------------------------------------------------------------------
// 3) QKV GEMM: [8192,192] x [192,3072], per-wave 16x64 register blocking.
//    q,k stored [B*H, N, 64]; v stored transposed [B*H, 64, N] (bf16).
// ---------------------------------------------------------------------------
__global__ __launch_bounds__(256) void qkv_gemm(
    const unsigned short* __restrict__ h, const unsigned short* __restrict__ wT,
    unsigned short* __restrict__ qw, unsigned short* __restrict__ kw,
    unsigned short* __restrict__ vtw) {
  int wave = threadIdx.x >> 5, lane = threadIdx.x & 31;
  int tile = blockIdx.x * 8 + wave;                  // 512 M-tiles * 48 N-groups
  int mt = tile / 48, ng = tile % 48;
  int c = lane & 15, hi = lane >> 4;
  int arow = mt * 16 + c;
  v8f acc[4] = {};
#pragma unroll
  for (int k0 = 0; k0 < DIM; k0 += 32) {
    FragB16 a;
#pragma unroll
    for (int j = 0; j < 8; ++j)
      a.u[j] = *(const unsigned int*)(h + (size_t)arow * DIM + k0 + a_kbase(j, hi));
#pragma unroll
    for (int t = 0; t < 4; ++t) {
      FragB16 b;
      int bcol = (ng * 4 + t) * 16 + c;
#pragma unroll
      for (int j = 0; j < 8; ++j)
        b.u[j] = *(const unsigned int*)(wT + (size_t)bcol * DIM + k0 + 16 * hi + 2 * j);
      acc[t] = __builtin_amdgcn_wmma_f32_16x16x32_bf16(false, a.v, false, b.v,
                                                       (short)0, acc[t], false, false);
    }
  }
  int nbase = ng * 64;
  int which = nbase >> 10;                           // 0=q 1=k 2=v
  int head  = (nbase & (INNER - 1)) >> 6;
#pragma unroll
  for (int t = 0; t < 4; ++t) {
    int d = t * 16 + c;
#pragma unroll
    for (int j = 0; j < 8; ++j) {
      int m   = mt * 16 + j + 8 * hi;
      int bb  = m >> 10;
      int tok = m & (NTOK - 1);
      int bh  = bb * HEADS + head;
      unsigned short val = f2bf(acc[t][j]);
      if (which == 0)      qw[((size_t)bh * NTOK + tok) * DHEAD + d] = val;
      else if (which == 1) kw[((size_t)bh * NTOK + tok) * DHEAD + d] = val;
      else                 vtw[((size_t)bh * DHEAD + d) * NTOK + tok] = val;
    }
  }
}

// ---------------------------------------------------------------------------
// 4) Flash attention: 4 waves/block, 16 query rows per wave, 64-key tiles.
//    K/V tiles staged once per block into LDS by the Tensor Data Mover,
//    double-buffered on TENSORcnt so DMA overlaps WMMA+softmax.
// ---------------------------------------------------------------------------
__global__ __launch_bounds__(128) void flash_attn(
    const unsigned short* __restrict__ qw, const unsigned short* __restrict__ kw,
    const unsigned short* __restrict__ vtw, unsigned short* __restrict__ attn) {
  __shared__ alignas(16) unsigned short kv_lds[2][2][64 * DHEAD]; // [buf][K/V] 8KB each
  __shared__ alignas(16) unsigned short plds[4][16 * 64];         // P tile per wave
  int wave = threadIdx.x >> 5, lane = threadIdx.x & 31;
  int tid = threadIdx.x;
  int c = lane & 15, hi = lane >> 4;
  int bh = blockIdx.x >> 4;                          // 0..127 (b*16+head)
  int qblk = blockIdx.x & 15;
  int q0 = qblk * 64 + wave * 16;
  const unsigned short* qp = qw  + (size_t)bh * NTOK * DHEAD;
  const unsigned short* kp = kw  + (size_t)bh * NTOK * DHEAD;
  const unsigned short* vp = vtw + (size_t)bh * DHEAD * NTOK;
  unsigned short* myp = plds[wave];

  FragB16 qa0, qa1;
#pragma unroll
  for (int j = 0; j < 8; ++j) {
    int kb = a_kbase(j, hi);
    qa0.u[j] = *(const unsigned int*)(qp + (size_t)(q0 + c) * DHEAD + kb);
    qa1.u[j] = *(const unsigned int*)(qp + (size_t)(q0 + c) * DHEAD + 32 + kb);
  }

  float mrun[8], lrun[8], alpha[8];
  v8f o[4] = {};
#pragma unroll
  for (int j = 0; j < 8; ++j) { mrun[j] = -1e30f; lrun[j] = 0.f; }

  // ---- KV tile staging (TDM DMA when available, cooperative loads else) ----
  auto stage = [&](int buf, int kt) {
#if USE_TDM
    if (wave == 0) {
      // K tile: 64 keys x 64 dims (row-major [tok][d], row stride 64)
      tdm_load_2d((unsigned)(size_t)&kv_lds[buf][0][0],
                  (unsigned long long)(size_t)(kp + (size_t)kt * DHEAD),
                  DHEAD, 64, DHEAD, NTOK, DHEAD);
      // V tile: 64 d-rows x 64 keys (row-major [d][tok], row stride 1024)
      tdm_load_2d((unsigned)(size_t)&kv_lds[buf][1][0],
                  (unsigned long long)(size_t)(vp + kt),
                  64, DHEAD, NTOK, DHEAD, NTOK);
    }
#else
#pragma unroll
    for (int i = 0; i < 4; ++i) {
      int idx = i * 128 + tid;                       // 512 x 16B chunks per tile
      int row = idx >> 3, seg = (idx & 7) * 8;
      *(uint4*)(&kv_lds[buf][0][row * DHEAD + seg]) =
          *(const uint4*)(kp + (size_t)(kt + row) * DHEAD + seg);
      *(uint4*)(&kv_lds[buf][1][row * DHEAD + seg]) =
          *(const uint4*)(vp + (size_t)row * NTOK + kt + seg);
    }
#endif
  };

  stage(0, 0);
  for (int kt = 0; kt < NTOK; kt += 64) {
    int cur = (kt >> 6) & 1;
    bool more = (kt + 64) < NTOK;
    if (more) stage(cur ^ 1, kt + 64);               // DMA next tile (async)
#if USE_TDM
    if (wave == 0) {
      if (more) __builtin_amdgcn_s_wait_tensorcnt(2);  // prev pair done, new pair in flight
      else      __builtin_amdgcn_s_wait_tensorcnt(0);
    }
#endif
    __syncthreads();
    const unsigned short* kbuf = &kv_lds[cur][0][0];
    const unsigned short* vbuf = &kv_lds[cur][1][0];

    v8f s[4] = {};
    FragB16 b;
    // ---- S = Q·K^T over 4 chunks of 16 keys (K read from LDS) ----
#pragma unroll
    for (int ch = 0; ch < 4; ++ch) {
      int krow = ch * 16 + c;
#pragma unroll
      for (int j = 0; j < 8; ++j)
        b.u[j] = *(const unsigned int*)(kbuf + krow * DHEAD + 16 * hi + 2 * j);
      s[ch] = __builtin_amdgcn_wmma_f32_16x16x32_bf16(false, qa0.v, false, b.v,
                                                      (short)0, s[ch], false, false);
#pragma unroll
      for (int j = 0; j < 8; ++j)
        b.u[j] = *(const unsigned int*)(kbuf + krow * DHEAD + 32 + 16 * hi + 2 * j);
      s[ch] = __builtin_amdgcn_wmma_f32_16x16x32_bf16(false, qa1.v, false, b.v,
                                                      (short)0, s[ch], false, false);
    }

    // ---- online softmax (per query row = per accumulator VGPR slot) ----
#pragma unroll
    for (int j = 0; j < 8; ++j) {
      float sc0 = s[0][j] * 0.125f;                  // DHEAD^-0.5
      float sc1 = s[1][j] * 0.125f;
      float sc2 = s[2][j] * 0.125f;
      float sc3 = s[3][j] * 0.125f;
      float mx = rmax16(fmaxf(fmaxf(sc0, sc1), fmaxf(sc2, sc3)));
      float mn = fmaxf(mrun[j], mx);
      alpha[j] = __expf(mrun[j] - mn);
      mrun[j] = mn;
      float p0 = __expf(sc0 - mn), p1 = __expf(sc1 - mn);
      float p2 = __expf(sc2 - mn), p3 = __expf(sc3 - mn);
      s[0][j] = p0; s[1][j] = p1; s[2][j] = p2; s[3][j] = p3;
      lrun[j] = lrun[j] * alpha[j] + rsum16((p0 + p1) + (p2 + p3));
    }
#pragma unroll
    for (int f = 0; f < 4; ++f)
#pragma unroll
      for (int j = 0; j < 8; ++j) o[f][j] *= alpha[j];

    // ---- P: C-fragment layout -> LDS -> two A-fragments (bf16) ----
#pragma unroll
    for (int j = 0; j < 8; ++j) {
      int r = (j + 8 * hi) * 64;
      myp[r + 0 * 16 + c] = f2bf(s[0][j]);
      myp[r + 1 * 16 + c] = f2bf(s[1][j]);
      myp[r + 2 * 16 + c] = f2bf(s[2][j]);
      myp[r + 3 * 16 + c] = f2bf(s[3][j]);
    }
    FragB16 pa0, pa1;
#pragma unroll
    for (int j = 0; j < 8; ++j) {
      int kb = a_kbase(j, hi);
      pa0.u[j] = *(const unsigned int*)(myp + c * 64 + kb);
      pa1.u[j] = *(const unsigned int*)(myp + c * 64 + 32 + kb);
    }

    // ---- O += P · V (V read from LDS, [d][key-local] layout) ----
#pragma unroll
    for (int f = 0; f < 4; ++f) {
#pragma unroll
      for (int j = 0; j < 8; ++j)
        b.u[j] = *(const unsigned int*)(vbuf + (f * 16 + c) * 64 + 16 * hi + 2 * j);
      o[f] = __builtin_amdgcn_wmma_f32_16x16x32_bf16(false, pa0.v, false, b.v,
                                                     (short)0, o[f], false, false);
#pragma unroll
      for (int j = 0; j < 8; ++j)
        b.u[j] = *(const unsigned int*)(vbuf + (f * 16 + c) * 64 + 32 + 16 * hi + 2 * j);
      o[f] = __builtin_amdgcn_wmma_f32_16x16x32_bf16(false, pa1.v, false, b.v,
                                                     (short)0, o[f], false, false);
    }
    __syncthreads();                                 // safe to overwrite other buffer
  }

  // ---- finalize and scatter to [B, N, INNER] bf16 for the output GEMM ----
  int bb = bh >> 4, head = bh & 15;
#pragma unroll
  for (int j = 0; j < 8; ++j) {
    float inv = 1.f / lrun[j];
    int row = q0 + j + 8 * hi;
    unsigned short* dst = attn + ((size_t)(bb * NTOK + row) * INNER + head * DHEAD);
    dst[0 * 16 + c] = f2bf(o[0][j] * inv);
    dst[1 * 16 + c] = f2bf(o[1][j] * inv);
    dst[2 * 16 + c] = f2bf(o[2][j] * inv);
    dst[3 * 16 + c] = f2bf(o[3][j] * inv);
  }
}

// ---------------------------------------------------------------------------
// 5) Output GEMM: [8192,1024] x [1024,192] + bias -> fp32 d_out.
// ---------------------------------------------------------------------------
__global__ __launch_bounds__(256) void out_gemm(
    const unsigned short* __restrict__ attn, const unsigned short* __restrict__ woT,
    const float* __restrict__ bo, float* __restrict__ out) {
  int wave = threadIdx.x >> 5, lane = threadIdx.x & 31;
  int tile = blockIdx.x * 8 + wave;                  // 512 M-tiles * 3 N-groups
  int mt = tile / 3, ng = tile % 3;
  int c = lane & 15, hi = lane >> 4;
  int arow = mt * 16 + c;
  v8f acc[4] = {};
  for (int k0 = 0; k0 < INNER; k0 += 32) {
    FragB16 a;
#pragma unroll
    for (int j = 0; j < 8; ++j)
      a.u[j] = *(const unsigned int*)(attn + (size_t)arow * INNER + k0 + a_kbase(j, hi));
#pragma unroll
    for (int t = 0; t < 4; ++t) {
      FragB16 b;
      int bcol = (ng * 4 + t) * 16 + c;
#pragma unroll
      for (int j = 0; j < 8; ++j)
        b.u[j] = *(const unsigned int*)(woT + (size_t)bcol * INNER + k0 + 16 * hi + 2 * j);
      acc[t] = __builtin_amdgcn_wmma_f32_16x16x32_bf16(false, a.v, false, b.v,
                                                       (short)0, acc[t], false, false);
    }
  }
#pragma unroll
  for (int t = 0; t < 4; ++t) {
    int ncol = (ng * 4 + t) * 16 + c;
    float bias = bo[ncol];
#pragma unroll
    for (int j = 0; j < 8; ++j) {
      int m = mt * 16 + j + 8 * hi;
      out[(size_t)m * DIM + ncol] = acc[t][j] + bias;
    }
  }
}

// ---------------------------------------------------------------------------
extern "C" void kernel_launch(void* const* d_in, const int* in_sizes, int n_in,
                              void* d_out, int out_size, void* d_ws, size_t ws_size,
                              hipStream_t stream) {
  (void)in_sizes; (void)n_in; (void)out_size; (void)ws_size;
  const float* x      = (const float*)d_in[0];
  const float* shift  = (const float*)d_in[1];
  const float* scale  = (const float*)d_in[2];
  const float* norm_w = (const float*)d_in[3];
  const float* norm_b = (const float*)d_in[4];
  const float* w_qkv  = (const float*)d_in[5];
  const float* w_out  = (const float*)d_in[6];
  const float* b_out  = (const float*)d_in[7];
  float* out = (float*)d_out;

  // Workspace carve (bf16 as u16), all offsets 4B-aligned.
  char* ws = (char*)d_ws;
  unsigned short* h_bf   = (unsigned short*)ws;  ws += (size_t)ROWS * DIM * 2;
  unsigned short* wqkvT  = (unsigned short*)ws;  ws += (size_t)3 * INNER * DIM * 2;
  unsigned short* woutT  = (unsigned short*)ws;  ws += (size_t)DIM * INNER * 2;
  unsigned short* qw     = (unsigned short*)ws;  ws += (size_t)BATCH * HEADS * NTOK * DHEAD * 2;
  unsigned short* kw     = (unsigned short*)ws;  ws += (size_t)BATCH * HEADS * NTOK * DHEAD * 2;
  unsigned short* vtw    = (unsigned short*)ws;  ws += (size_t)BATCH * HEADS * DHEAD * NTOK * 2;
  unsigned short* attn   = (unsigned short*)ws;  ws += (size_t)ROWS * INNER * 2;

  conv_wqkv<<<(3 * INNER * DIM + 255) / 256, 256, 0, stream>>>(w_qkv, wqkvT);
  conv_wout<<<(DIM * INNER + 255) / 256, 256, 0, stream>>>(w_out, woutT);
  ln_mod<<<ROWS / 8, 256, 0, stream>>>(x, shift, scale, norm_w, norm_b, h_bf);
  qkv_gemm<<<(ROWS / 16) * 48 / 8, 256, 0, stream>>>(h_bf, wqkvT, qw, kw, vtw);
  flash_attn<<<BATCH * HEADS * (NTOK / 64), 128, 0, stream>>>(qw, kw, vtw, attn);
  out_gemm<<<(ROWS / 16) * 3 / 8, 256, 0, stream>>>(attn, woutT, b_out, out);
}
